// MoChA_41910290874403
// MI455X (gfx1250) — matrix operand order repaired and checked by reading, not compiled, not run on hardware
//
#include <hip/hip_runtime.h>
#include <hip/hip_bf16.h>

// ---------------------------------------------------------------------------
// MoChA forward for MI455X (gfx1250): bf16 WMMA GEMMs + async global->LDS
// tile staging + LDS block-scan kernels
// ---------------------------------------------------------------------------

typedef __bf16 bf16_t;
typedef __bf16 v16bf __attribute__((ext_vector_type(16)));
typedef float  v8f   __attribute__((ext_vector_type(8)));

#if __has_builtin(__builtin_amdgcn_global_load_async_to_lds_b64)
#define USE_ASYNC_LDS 1
// Parameter types per clang diagnostic: v2i in AS1 (global) / AS3 (LDS)
typedef int v2i_t __attribute__((vector_size(2 * sizeof(int))));
typedef __attribute__((address_space(1))) v2i_t* g_v2i_p;
typedef __attribute__((address_space(3))) v2i_t* l_v2i_p;
#endif

constexpr int cBATCH = 16;
constexpr int cKLEN  = 1500;
constexpr int cQLEN  = 200;
constexpr int cADIM  = 512;
constexpr int cHMA   = 4;
constexpr int cHCA   = 2;
constexpr int cHTOT  = 8;
constexpr int cWIN   = 4;
constexpr float cEPS = 1e-6f;
constexpr float cNEG = -1000000000.0f;

union Pack4 {            // 4 bf16 <-> 8 bytes
    bf16_t h[4];
    uint2  u;
};
struct Frag32 {          // 32 bytes -> v16bf
    uint4 lo, hi;
};

// ---------------------------------------------------------------------------
// Async global->LDS copy of 8 bytes (one b64 per lane), with fallback.
// ---------------------------------------------------------------------------
__device__ __forceinline__ void async_copy_b64(const bf16_t* gsrc, bf16_t* ldst) {
#ifdef USE_ASYNC_LDS
    __builtin_amdgcn_global_load_async_to_lds_b64(
        (g_v2i_p)gsrc, (l_v2i_p)ldst, 0, 0);
#else
    *(uint2*)ldst = *(const uint2*)gsrc;
#endif
}

__device__ __forceinline__ void wait_async_lds() {
#ifdef USE_ASYNC_LDS
#if __has_builtin(__builtin_amdgcn_s_wait_asynccnt)
    __builtin_amdgcn_s_wait_asynccnt(0);
#else
    asm volatile("s_wait_asynccnt 0x0" ::: "memory");
#endif
#endif
}

// ---------------------------------------------------------------------------
// f32 -> bf16 conversion, 4 elements per thread (sizes are multiples of 4)
// ---------------------------------------------------------------------------
__global__ __launch_bounds__(256) void f32_to_bf16_kernel(const float* __restrict__ x,
                                                          bf16_t* __restrict__ y,
                                                          long long n4) {
    long long i = (long long)blockIdx.x * blockDim.x + threadIdx.x;
    long long stride = (long long)gridDim.x * blockDim.x;
    for (; i < n4; i += stride) {
        float4 f = ((const float4*)x)[i];
        Pack4 p;
        p.h[0] = (bf16_t)f.x; p.h[1] = (bf16_t)f.y;
        p.h[2] = (bf16_t)f.z; p.h[3] = (bf16_t)f.w;
        ((uint2*)y)[i] = p.u;
    }
}

// ---------------------------------------------------------------------------
// LDS fragment loaders: 2x ds_load_b128 each, per ISA 7.12.2 bf16 layouts.
// ---------------------------------------------------------------------------
__device__ __forceinline__ v16bf load_frag_a(const bf16_t* row, int hi) {
    Frag32 f;
    f.lo = *(const uint4*)(row + hi * 8);
    f.hi = *(const uint4*)(row + 16 + hi * 8);
    return __builtin_bit_cast(v16bf, f);
}
__device__ __forceinline__ v16bf load_frag_b(const bf16_t* row, int hi) {
    Frag32 f;
    f.lo = *(const uint4*)(row + hi * 8);
    f.hi = *(const uint4*)(row + 16 + hi * 8);
    return __builtin_bit_cast(v16bf, f);
}

// ---------------------------------------------------------------------------
// Generic batched WMMA GEMM:  C[b,h][m,n] = scale * (A @ B) + addp[0] + bias[n]
// Block: 256 threads = 8 waves; tile 64(M) x 128(N); wave grid 2x4; each wave
// computes a 32x32 macro-tile (4 accumulators) -> 4 WMMA per K-step.
// Tile staging: async global->LDS b64 on full K-slabs; manual (branchless,
// clamped + select-zeroed) path on the K tail and for the transposing B path.
// ---------------------------------------------------------------------------
__global__ __launch_bounds__(256) void wmma_gemm_bf16(
    const bf16_t* __restrict__ A, const bf16_t* __restrict__ Bm,
    const float* __restrict__ bias, const float* __restrict__ addp,
    float* __restrict__ Cf, bf16_t* __restrict__ Cb,
    int M, int N, int K, int H,
    long long aB, long long aH, long long aM,
    long long bB, long long bH, long long bK, long long bN,
    long long cB, long long cH, long long cM,
    float scale,
    const unsigned char* __restrict__ mask, long long mB, long long mM) {

    __shared__ __align__(16) bf16_t As[64][40];     // 64 rows  x 32 K, pitch 80B
    __shared__ __align__(16) bf16_t BsT[128][40];   // 128 cols x 32 K, pitch 80B

    const int tid  = threadIdx.x;
    const int lane = tid & 31;
    const int wave = tid >> 5;
    const int wr   = wave >> 2;      // 0..1 : 32-row band
    const int wn   = wave & 3;       // 0..3 : 32-col band
    const int hi   = lane >> 4;
    const int l16  = lane & 15;

    const int z  = blockIdx.z;
    const int bb = z / H;
    const int hh = z % H;

    const bf16_t* Ab = A  + (long long)bb * aB + (long long)hh * aH;
    const bf16_t* Bb = Bm + (long long)bb * bB + (long long)hh * bH;
    const long long cbase = (long long)bb * cB + (long long)hh * cH;

    const int m0 = blockIdx.y * 64;
    const int n0 = blockIdx.x * 128;

    v8f acc[2][2] = {{v8f{}, v8f{}}, {v8f{}, v8f{}}};

    for (int k0 = 0; k0 < K; k0 += 32) {
        const bool ktail = (k0 + 32) > K;   // uniform

        // ---- A tile: 64x32 (2 groups of 4 bf16 per thread), branchless ------
        #pragma unroll
        for (int i = 0; i < 2; ++i) {
            int g  = tid + i * 256;
            int r  = g >> 3;
            int kq = (g & 7) * 4;
            int gm = m0 + r;   gm = (gm < M) ? gm : (M - 1);       // clamp
            int gk = k0 + kq;
            const bf16_t* src = Ab + (long long)gm * aM + gk;
            if (!ktail) {
                async_copy_b64(src, &As[r][kq]);
            } else {
                Pack4 p;
                p.u = *(const uint2*)src;
                #pragma unroll
                for (int t = 0; t < 4; ++t)
                    p.h[t] = (gk + t < K) ? p.h[t] : (bf16_t)0.0f;
                *(uint2*)&As[r][kq] = p.u;
            }
        }
        // ---- B tile: 128 cols x 32 K, staged transposed ---------------------
        if (bK == 1) {
            // k-contiguous in memory: vector copy along K
            #pragma unroll
            for (int i = 0; i < 4; ++i) {
                int g  = tid + i * 256;
                int n  = g >> 3;
                int kq = (g & 7) * 4;
                int gn = n0 + n;   gn = (gn < N) ? gn : (N - 1);   // clamp
                int gk = k0 + kq;
                const bf16_t* src = Bb + gk + (long long)gn * bN;
                if (!ktail) {
                    async_copy_b64(src, &BsT[n][kq]);
                } else {
                    Pack4 p;
                    p.u = *(const uint2*)src;
                    #pragma unroll
                    for (int t = 0; t < 4; ++t)
                        p.h[t] = (gk + t < K) ? p.h[t] : (bf16_t)0.0f;
                    *(uint2*)&BsT[n][kq] = p.u;
                }
            }
        } else {
            // n-contiguous in memory (bN==1): vector load along N, transpose
            #pragma unroll
            for (int i = 0; i < 4; ++i) {
                int g  = tid + i * 256;
                int kk = g >> 5;
                int nq = (g & 31) * 4;
                int gk = k0 + kk;
                int gkc = (gk < K) ? gk : (K - 1);                  // clamp
                int gn = n0 + nq;  gn = (gn + 3 < N) ? gn : (N - 4); // clamp
                Pack4 p;
                p.u = *(const uint2*)(Bb + (long long)gkc * bK + gn);
                if (ktail) {
                    p.h[0] = (gk < K) ? p.h[0] : (bf16_t)0.0f;
                    p.h[1] = (gk < K) ? p.h[1] : (bf16_t)0.0f;
                    p.h[2] = (gk < K) ? p.h[2] : (bf16_t)0.0f;
                    p.h[3] = (gk < K) ? p.h[3] : (bf16_t)0.0f;
                }
                #pragma unroll
                for (int t = 0; t < 4; ++t) BsT[nq + t][kk] = p.h[t];
            }
        }
        // speculative prefetch of next A K-slab (global_prefetch_b8)
        if (k0 + 32 < K) {
            __builtin_prefetch(Ab + (long long)(m0 + (tid & 63)) * aM + (k0 + 32), 0, 1);
        }
        wait_async_lds();
        __syncthreads();

        v16bf a0 = load_frag_a(&As[wr * 32 + l16][0], hi);
        v16bf a1 = load_frag_a(&As[wr * 32 + 16 + l16][0], hi);
        v16bf b0 = load_frag_b(&BsT[wn * 32 + l16][0], hi);
        v16bf b1 = load_frag_b(&BsT[wn * 32 + 16 + l16][0], hi);

        acc[0][0] = __builtin_amdgcn_wmma_f32_16x16x32_bf16(
            false, a0, false, b0, (short)0, acc[0][0], false, false);
        acc[0][1] = __builtin_amdgcn_wmma_f32_16x16x32_bf16(
            false, a0, false, b1, (short)0, acc[0][1], false, false);
        acc[1][0] = __builtin_amdgcn_wmma_f32_16x16x32_bf16(
            false, a1, false, b0, (short)0, acc[1][0], false, false);
        acc[1][1] = __builtin_amdgcn_wmma_f32_16x16x32_bf16(
            false, a1, false, b1, (short)0, acc[1][1], false, false);
        __syncthreads();
    }

    // ---- epilogue -----------------------------------------------------------
    const float addv = addp ? addp[0] : 0.0f;
    const bool  mfull = (m0 + 63) < M;     // uniform
    #pragma unroll
    for (int sn = 0; sn < 2; ++sn) {
        const int ncol = n0 + wn * 32 + sn * 16 + l16;
        if (ncol < N) {
            const float bterm = (bias ? bias[ncol] : 0.0f) + addv;
            #pragma unroll
            for (int sm = 0; sm < 2; ++sm) {
                const int mbase = m0 + wr * 32 + sm * 16 + (hi ? 8 : 0);
                if (mfull) {
                    #pragma unroll
                    for (int r2 = 0; r2 < 8; ++r2) {
                        const int mrow = mbase + r2;
                        float v = acc[sm][sn][r2] * scale + bterm;
                        if (mask && !mask[(long long)bb * mB + (long long)mrow * mM + ncol])
                            v = cNEG;
                        long long ci = cbase + (long long)mrow * cM + ncol;
                        if (Cf) Cf[ci] = v;
                        else    Cb[ci] = (bf16_t)v;
                    }
                } else {
                    #pragma unroll
                    for (int r2 = 0; r2 < 8; ++r2) {
                        const int mrow = mbase + r2;
                        if (mrow < M) {
                            float v = acc[sm][sn][r2] * scale + bterm;
                            if (mask && !mask[(long long)bb * mB + (long long)mrow * mM + ncol])
                                v = cNEG;
                            long long ci = cbase + (long long)mrow * cM + ncol;
                            if (Cf) Cf[ci] = v;
                            else    Cb[ci] = (bf16_t)v;
                        }
                    }
                }
            }
        }
    }
}

// ---------------------------------------------------------------------------
// Block-wide exclusive scan helper (256 threads, Hillis-Steele in LDS)
// ---------------------------------------------------------------------------
__device__ __forceinline__ float block_excl_scan(float v, float* red) {
    int t = threadIdx.x;
    red[t] = v;
    __syncthreads();
    #pragma unroll
    for (int off = 1; off < 256; off <<= 1) {
        float add = (t >= off) ? red[t - off] : 0.0f;
        __syncthreads();
        red[t] += add;
        __syncthreads();
    }
    float incl = red[t];
    __syncthreads();
    return incl - v;
}

// ---------------------------------------------------------------------------
// Monotonic attention scan: one block per (b, h_ma); sequential over q.
// ---------------------------------------------------------------------------
constexpr int SCAN_CH = 6;  // 256 * 6 = 1536 >= KLEN

__global__ __launch_bounds__(256) void mocha_scan_kernel(const float* __restrict__ e_ma,
                                                         float* __restrict__ alpha) {
    __shared__ float red[256];
    const long long zoff = (long long)blockIdx.x * cQLEN * cKLEN;
    const float* e = e_ma + zoff;
    float* al = alpha + zoff;

    const int base = threadIdx.x * SCAN_CH;
    float aw[SCAN_CH];
    #pragma unroll
    for (int i = 0; i < SCAN_CH; ++i) aw[i] = (base + i == 0) ? 1.0f : 0.0f;

    for (int q = 0; q < cQLEN; ++q) {
        const float* er = e + (long long)q * cKLEN;
        float p[SCAN_CH], tpre[SCAN_CH], cp[SCAN_CH];

        float run = 0.0f;
        #pragma unroll
        for (int i = 0; i < SCAN_CH; ++i) {
            int k = base + i;
            float ei = (k < cKLEN) ? er[k] : cNEG;
            p[i] = 1.0f / (1.0f + __expf(-ei));
            float om = 1.0f - p[i];
            om = fminf(fmaxf(om, cEPS), 1.0f);
            run += __logf(om);
            tpre[i] = run;
        }
        float ex = block_excl_scan(run, red);
        #pragma unroll
        for (int i = 0; i < SCAN_CH; ++i)
            cp[i] = __expf((i == 0) ? ex : (ex + tpre[i - 1]));

        float lrun = 0.0f;
        float lpre[SCAN_CH];
        #pragma unroll
        for (int i = 0; i < SCAN_CH; ++i) {
            float den = fminf(fmaxf(cp[i], cEPS), 1.0f);
            lrun += aw[i] / den;
            lpre[i] = lrun;
        }
        float ex2 = block_excl_scan(lrun, red);

        #pragma unroll
        for (int i = 0; i < SCAN_CH; ++i) {
            float awn = p[i] * cp[i] * (ex2 + lpre[i]);
            aw[i] = awn;
            int k = base + i;
            if (k < cKLEN) al[(long long)q * cKLEN + k] = awn;
        }
    }
}

// ---------------------------------------------------------------------------
// Beta: one block per (b*H_TOT + h, q) row.
// ---------------------------------------------------------------------------
__global__ __launch_bounds__(256) void beta_kernel(const float* __restrict__ e_ca,
                                                   const float* __restrict__ alpha,
                                                   bf16_t* __restrict__ beta) {
    __shared__ float sexp[cKLEN];
    __shared__ float ratio[cKLEN + cWIN];
    __shared__ float red[256];

    const int q = blockIdx.x;
    const int z = blockIdx.y;            // b*H_TOT + h
    const int b = z / cHTOT;
    const int h = z % cHTOT;
    const int hma = h / cHCA;

    const long long erow = ((long long)z * cQLEN + q) * (long long)cKLEN;
    const float* er = e_ca + erow;
    const float* ar = alpha + (((long long)(b * cHMA + hma) * cQLEN) + q) * (long long)cKLEN;

    const int tid = threadIdx.x;

    float mx = -1e30f;
    for (int k = tid; k < cKLEN; k += 256) mx = fmaxf(mx, er[k]);
    red[tid] = mx;
    __syncthreads();
    for (int off = 128; off > 0; off >>= 1) {
        if (tid < off) red[tid] = fmaxf(red[tid], red[tid + off]);
        __syncthreads();
    }
    mx = red[0];
    __syncthreads();

    for (int k = tid; k < cKLEN; k += 256)
        sexp[k] = fmaxf(__expf(er[k] - mx), 1e-5f);
    if (tid < cWIN) ratio[cKLEN + tid] = 0.0f;
    __syncthreads();

    for (int k = tid; k < cKLEN; k += 256) {
        float den = 0.0f;
        int jlo = (k - (cWIN - 1) > 0) ? k - (cWIN - 1) : 0;
        for (int j = jlo; j <= k; ++j) den += sexp[j];
        ratio[k] = ar[k] / den;
    }
    __syncthreads();

    for (int k = tid; k < cKLEN; k += 256) {
        float s = 0.0f;
        #pragma unroll
        for (int j = 0; j < cWIN; ++j) s += ratio[k + j];  // padded tail is 0
        beta[erow + k] = (bf16_t)(sexp[k] * s);
    }
}

// ---------------------------------------------------------------------------
// Host launch
// ---------------------------------------------------------------------------
extern "C" void kernel_launch(void* const* d_in, const int* in_sizes, int n_in,
                              void* d_out, int out_size, void* d_ws, size_t ws_size,
                              hipStream_t stream) {
    (void)in_sizes; (void)n_in; (void)out_size; (void)ws_size;

    const float* key   = (const float*)d_in[0];
    const float* value = (const float*)d_in[1];
    const float* query = (const float*)d_in[2];
    const unsigned char* mask = (const unsigned char*)d_in[3];
    const float* Wk_ma = (const float*)d_in[4];
    const float* bk_ma = (const float*)d_in[5];
    const float* Wq_ma = (const float*)d_in[6];
    const float* bq_ma = (const float*)d_in[7];
    const float* rptr  = (const float*)d_in[8];
    const float* Wk_ca = (const float*)d_in[9];
    const float* bk_ca = (const float*)d_in[10];
    const float* Wq_ca = (const float*)d_in[11];
    const float* bq_ca = (const float*)d_in[12];
    const float* Wv    = (const float*)d_in[13];
    const float* bv    = (const float*)d_in[14];
    const float* Wout  = (const float*)d_in[15];
    const float* bout  = (const float*)d_in[16];
    float* out = (float*)d_out;

    const long long MK  = (long long)cBATCH * cKLEN;   // 24000
    const long long MQ  = (long long)cBATCH * cQLEN;   // 3200
    const long long WSZ = (long long)cADIM * cADIM;    // 262144

    size_t off = 0;
    auto bump = [&](size_t bytes) -> char* {
        char* p = (char*)d_ws + off;
        off = (off + bytes + 255) & ~(size_t)255;
        return p;
    };
    bf16_t* key_bf   = (bf16_t*)bump(MK * cADIM * 2);
    bf16_t* val_bf   = (bf16_t*)bump(MK * cADIM * 2);
    bf16_t* qry_bf   = (bf16_t*)bump(MQ * cADIM * 2);
    bf16_t* Wkma_bf  = (bf16_t*)bump(WSZ * 2);
    bf16_t* Wqma_bf  = (bf16_t*)bump(WSZ * 2);
    bf16_t* Wkca_bf  = (bf16_t*)bump(WSZ * 2);
    bf16_t* Wqca_bf  = (bf16_t*)bump(WSZ * 2);
    bf16_t* Wv_bf    = (bf16_t*)bump(WSZ * 2);
    bf16_t* Wout_bf  = (bf16_t*)bump(WSZ * 2);
    bf16_t* kma_bf   = (bf16_t*)bump(MK * cADIM * 2);
    bf16_t* qma_bf   = (bf16_t*)bump(MQ * cADIM * 2);
    bf16_t* kca_bf   = (bf16_t*)bump(MK * cADIM * 2);
    bf16_t* qca_bf   = (bf16_t*)bump(MQ * cADIM * 2);
    bf16_t* v_bf     = (bf16_t*)bump(MK * cADIM * 2);
    float*  e_ma     = (float*)bump((long long)cBATCH * cHMA * cQLEN * cKLEN * 4);
    float*  alpha    = (float*)bump((long long)cBATCH * cHMA * cQLEN * cKLEN * 4);
    float*  e_ca     = (float*)bump((long long)cBATCH * cHTOT * cQLEN * cKLEN * 4);
    bf16_t* beta_bf  = (bf16_t*)bump((long long)cBATCH * cHTOT * cQLEN * cKLEN * 2);
    bf16_t* cv_bf    = (bf16_t*)bump(MQ * cADIM * 2);

    auto conv = [&](const float* src, bf16_t* dst, long long n) {
        f32_to_bf16_kernel<<<2048, 256, 0, stream>>>(src, dst, n / 4);
    };
    conv(key,   key_bf,  MK * cADIM);
    conv(value, val_bf,  MK * cADIM);
    conv(query, qry_bf,  MQ * cADIM);
    conv(Wk_ma, Wkma_bf, WSZ);
    conv(Wq_ma, Wqma_bf, WSZ);
    conv(Wk_ca, Wkca_bf, WSZ);
    conv(Wq_ca, Wqca_bf, WSZ);
    conv(Wv,    Wv_bf,   WSZ);
    conv(Wout,  Wout_bf, WSZ);

    auto gemm = [&](const bf16_t* A, const bf16_t* Bm, const float* bias,
                    const float* addp, float* Cf, bf16_t* Cb,
                    int M, int N, int K, int H, int batches,
                    long long aB, long long aH, long long aM,
                    long long bB, long long bH, long long bK, long long bN,
                    long long cB, long long cH, long long cM,
                    float scale, const unsigned char* msk,
                    long long mB, long long mM) {
        dim3 grid((N + 127) / 128, (M + 63) / 64, batches);
        wmma_gemm_bf16<<<grid, 256, 0, stream>>>(
            A, Bm, bias, addp, Cf, Cb, M, N, K, H,
            aB, aH, aM, bB, bH, bK, bN, cB, cH, cM, scale, msk, mB, mM);
    };

    const long long KA = (long long)cKLEN * cADIM;  // 768000
    const long long QA = (long long)cQLEN * cADIM;  // 102400
    const long long QK = (long long)cQLEN * cKLEN;  // 300000

    // ---- projections (bias, bf16 out) ---------------------------------------
    gemm(key_bf, Wkma_bf, bk_ma, nullptr, nullptr, kma_bf,
         (int)MK, cADIM, cADIM, 1, 1, 0, 0, cADIM, 0, 0, cADIM, 1, 0, 0, cADIM,
         1.0f, nullptr, 0, 0);
    gemm(qry_bf, Wqma_bf, bq_ma, nullptr, nullptr, qma_bf,
         (int)MQ, cADIM, cADIM, 1, 1, 0, 0, cADIM, 0, 0, cADIM, 1, 0, 0, cADIM,
         1.0f, nullptr, 0, 0);
    gemm(key_bf, Wkca_bf, bk_ca, nullptr, nullptr, kca_bf,
         (int)MK, cADIM, cADIM, 1, 1, 0, 0, cADIM, 0, 0, cADIM, 1, 0, 0, cADIM,
         1.0f, nullptr, 0, 0);
    gemm(qry_bf, Wqca_bf, bq_ca, nullptr, nullptr, qca_bf,
         (int)MQ, cADIM, cADIM, 1, 1, 0, 0, cADIM, 0, 0, cADIM, 1, 0, 0, cADIM,
         1.0f, nullptr, 0, 0);
    gemm(val_bf, Wv_bf, bv, nullptr, nullptr, v_bf,
         (int)MK, cADIM, cADIM, 1, 1, 0, 0, cADIM, 0, 0, cADIM, 1, 0, 0, cADIM,
         1.0f, nullptr, 0, 0);

    // ---- e_ma = q_ma . k_ma^T / sqrt(128) + r, masked  (f32 out) ------------
    gemm(qma_bf, kma_bf, nullptr, rptr, e_ma, nullptr,
         cQLEN, cKLEN, 128, cHMA, cBATCH * cHMA,
         QA, 128, cADIM,
         KA, 128, 1, cADIM,
         (long long)cHMA * QK, QK, cKLEN,
         0.08838834764831845f, mask, QK, cKLEN);

    // ---- monotonic scan -> alpha --------------------------------------------
    mocha_scan_kernel<<<cBATCH * cHMA, 256, 0, stream>>>(e_ma, alpha);

    // ---- e_ca = q_ca . k_ca^T / 8, masked  (f32 out) ------------------------
    gemm(qca_bf, kca_bf, nullptr, nullptr, e_ca, nullptr,
         cQLEN, cKLEN, 64, cHTOT, cBATCH * cHTOT,
         QA, 64, cADIM,
         KA, 64, 1, cADIM,
         (long long)cHTOT * QK, QK, cKLEN,
         0.125f, mask, QK, cKLEN);

    // ---- beta (windowed chunkwise weights, bf16 out) ------------------------
    {
        dim3 grid(cQLEN, cBATCH * cHTOT);
        beta_kernel<<<grid, 256, 0, stream>>>(e_ca, alpha, beta_bf);
    }

    // ---- cv = beta . v  (bf16 out, laid out (b,q,h*64+d)) -------------------
    gemm(beta_bf, v_bf, nullptr, nullptr, nullptr, cv_bf,
         cQLEN, 64, cKLEN, cHTOT, cBATCH * cHTOT,
         (long long)cHTOT * QK, QK, cKLEN,
         KA, 64, cADIM, 1,
         QA, 64, cADIM,
         1.0f, nullptr, 0, 0);

    // ---- out = cv . Wout + bout  (f32 -> d_out) -----------------------------
    gemm(cv_bf, Wout_bf, bout, nullptr, out, nullptr,
         (int)MQ, cADIM, cADIM, 1, 1, 0, 0, cADIM, 0, 0, cADIM, 1, 0, 0, cADIM,
         1.0f, nullptr, 0, 0);
}
